// AttentionDTI_49323404427631
// MI455X (gfx1250) — matrix operand order repaired and checked by reading, not compiled, not run on hardware
//
#include <hip/hip_runtime.h>
#include <hip/hip_bf16.h>

typedef __attribute__((ext_vector_type(16))) _Float16 v16h;
typedef __attribute__((ext_vector_type(8)))  _Float16 v8h;
typedef __attribute__((ext_vector_type(8)))  float    v8f;

#define DIM   64
#define CONV  40
#define C2    80
#define C4    160
#define NB    8
#define LP0   100
#define LP1   97
#define LP2   92
#define LP3   85
#define LM0   1000
#define LM1   997
#define LM2   992
#define LM3   985

// ---------------------------------------------------------------------------
// Embedding gather: out[b,l,c] = (f16) emb[idx[b,l]*64 + c]
// ---------------------------------------------------------------------------
__global__ void dti_embed(const int* __restrict__ idx, const float* __restrict__ emb,
                          _Float16* __restrict__ out, int total /* B*L */) {
    int t = blockIdx.x * blockDim.x + threadIdx.x;
    if (t >= total * DIM) return;
    int pos = t >> 6, c = t & 63;
    out[t] = (_Float16)emb[idx[pos] * DIM + c];
}

// ---------------------------------------------------------------------------
// Fragment-order weight packing.
// Packed layout: block (nt, kt) at ((nt*kTiles)+kt)*512 elements; within a
// block lane l owns 16 consecutive f16 at l*16: its WMMA B fragment
// (col = nt*16 + (l&15), k = kt*32 + (l>>4)*16 + i). Zero-padded past K/N, so
// the GEMM needs no bounds checks on B.
// ---------------------------------------------------------------------------
__global__ void dti_pack_frag_conv(const float* __restrict__ w, _Float16* __restrict__ out,
                                   int Cout, int Cin, int Kk, int kTiles, int nTiles) {
    int t = blockIdx.x * blockDim.x + threadIdx.x;
    if (t >= kTiles * nTiles * 512) return;
    int e    = t & 511;
    int blk  = t >> 9;
    int kt   = blk % kTiles;
    int nt   = blk / kTiles;
    int lane = e >> 4;
    int i    = e & 15;
    int k    = kt * 32 + ((lane >> 4) << 4) + i;   // logical K index (= kk*Cin+cin)
    int col  = nt * 16 + (lane & 15);              // logical N index (= cout)
    float v = 0.f;
    if (k < Kk * Cin && col < Cout) {
        int kk = k / Cin, cin = k % Cin;
        v = w[(col * Cin + cin) * Kk + kk];
    }
    out[t] = (_Float16)v;
}

__global__ void dti_pack_frag_mat(const float* __restrict__ W, _Float16* __restrict__ out,
                                  int K, int N, int kTiles, int nTiles) {
    int t = blockIdx.x * blockDim.x + threadIdx.x;
    if (t >= kTiles * nTiles * 512) return;
    int e    = t & 511;
    int blk  = t >> 9;
    int kt   = blk % kTiles;
    int nt   = blk / kTiles;
    int lane = e >> 4;
    int i    = e & 15;
    int k    = kt * 32 + ((lane >> 4) << 4) + i;
    int col  = nt * 16 + (lane & 15);
    float v = 0.f;
    if (k < K && col < N) v = W[(long)k * N + col];
    out[t] = (_Float16)v;
}

// ---------------------------------------------------------------------------
// f32 -> f16 convert
// ---------------------------------------------------------------------------
__global__ void dti_cvt_h(const float* __restrict__ in, _Float16* __restrict__ out, long n) {
    long t = (long)blockIdx.x * blockDim.x + threadIdx.x;
    if (t < n) out[t] = (_Float16)in[t];
}

// ---------------------------------------------------------------------------
// Single-wave WMMA GEMM, one 16x16 tile per block (wave32).
//   D[m,n] = act( sum_k A[rowStart(m)+k] * Blogical[k,n] + bias[n] )
//   rowStart(m) = (m/Lout)*batchStride + (m%Lout)*rowStride  (im2col rows are
//   contiguous slices). All A/B fragment loads are unguarded 16-byte loads:
//   B is fragment-packed + zero-padded; A K-tail garbage multiplies zeros and
//   activation buffers carry +64 elements of slack.
// act: 0=none, 1=relu, 2=leaky(0.01)
// ---------------------------------------------------------------------------
__global__ void dti_wmma_gemm(const _Float16* __restrict__ A,
                              const _Float16* __restrict__ Bfrag,
                              const float* __restrict__ bias,
                              _Float16* __restrict__ outH,
                              float* __restrict__ outF,
                              int M, int N, int kTiles,
                              int Lout, int rowStride, int batchStride,
                              int act) {
    const int lane  = threadIdx.x;        // 0..31
    const int hi    = lane >> 4;          // 0 or 1
    const int tileM = blockIdx.y << 4;
    const int tileN = blockIdx.x << 4;

    const int m = tileM + (lane & 15);    // A-matrix row owned by this lane
    long aBase = 0;
    if (m < M) aBase = (long)(m / Lout) * batchStride + (long)(m % Lout) * rowStride;
    const _Float16* __restrict__ ap = A + aBase + (hi << 4);

    // per-lane B fragment stream: 16 f16 per 512-element block
    const v8h* __restrict__ bp =
        (const v8h*)(Bfrag + ((long)blockIdx.x * kTiles) * 512 + lane * 16);

    v8f acc = {};
    for (int kt = 0; kt < kTiles; ++kt) {
        const int k0 = kt << 5;
        v8h a0 = *(const v8h*)(ap + k0);        // contiguous K half: k = 16*hi + s
        v8h a1 = *(const v8h*)(ap + k0 + 8);
        v8h b0 = bp[0];
        v8h b1 = bp[1];
        bp += 64;                               // next 512-element block
        __builtin_prefetch(bp, 0, 1);
        v16h a = __builtin_shufflevector(a0, a1, 0, 1, 2, 3, 4, 5, 6, 7,
                                         8, 9, 10, 11, 12, 13, 14, 15);
        v16h b = __builtin_shufflevector(b0, b1, 0, 1, 2, 3, 4, 5, 6, 7,
                                         8, 9, 10, 11, 12, 13, 14, 15);
        acc = __builtin_amdgcn_wmma_f32_16x16x32_f16(false, a, false, b,
                                                     (short)0, acc, false, false);
    }

    const int col = tileN + (lane & 15);
    if (col < N) {
        const float bv = bias ? bias[col] : 0.f;
        #pragma unroll
        for (int j = 0; j < 8; ++j) {
            int mm = tileM + j + (hi << 3);     // C layout: M = j + 8*hi
            if (mm < M) {
                float v = acc[j] + bv;
                if (act == 1)      v = v > 0.f ? v : 0.f;
                else if (act == 2) v = v > 0.f ? v : 0.01f * v;
                if (outF) outF[(long)mm * N + col] = v;
                if (outH) outH[(long)mm * N + col] = (_Float16)v;
            }
        }
    }
}

// ---------------------------------------------------------------------------
// S[b,i,c] = mean_j relu(Aatt[b,i,c] + Batt[b,j,c])
// (mean commuted through the Wa linear: kills the [8,85,985,160] tensor GEMM)
// grid = (LA, B), block = C (160 = 5 waves)
// ---------------------------------------------------------------------------
__global__ void dti_cross_relu_mean(const float* __restrict__ Aatt,
                                    const float* __restrict__ Batt,
                                    float* __restrict__ S,
                                    int LA, int LB, int C) {
    int b = blockIdx.y, i = blockIdx.x, c = threadIdx.x;
    if (c >= C) return;
    float a = Aatt[((long)b * LA + i) * C + c];
    const float* bp = Batt + (long)b * LB * C + c;
    float s = 0.f;
    for (int j = 0; j < LB; ++j) {
        float v = a + bp[(long)j * C];
        s += (v > 0.f) ? v : 0.f;
    }
    S[((long)b * LA + i) * C + c] = s * (1.f / (float)LB);
}

// ---------------------------------------------------------------------------
// pair[b, pairOff+c] = max_l  X[b,l,c] * (0.5 + sigmoid(G[b,l,c]))
// ---------------------------------------------------------------------------
__global__ void dti_gate_max(const _Float16* __restrict__ X,
                             const float* __restrict__ G,
                             _Float16* __restrict__ pair,
                             int L, int C, int pairStride, int pairOff) {
    int b = blockIdx.x, c = threadIdx.x;
    if (c >= C) return;
    float best = -3.4e38f;
    const _Float16* xp = X + (long)b * L * C + c;
    const float*    gp = G + (long)b * L * C + c;
    for (int l = 0; l < L; ++l) {
        float x = (float)xp[(long)l * C];
        float s = 1.f / (1.f + expf(-gp[(long)l * C]));
        float v = x * (0.5f + s);
        best = best > v ? best : v;
    }
    pair[(long)b * pairStride + pairOff + c] = (_Float16)best;
}

// ---------------------------------------------------------------------------
// Host side
// ---------------------------------------------------------------------------
static inline void* wsTake(char*& p, size_t elems, size_t elemBytes) {
    void* r = (void*)p;
    size_t bytes = ((elems + 64) * elemBytes + 255) & ~((size_t)255); // +64 elem slack
    p += bytes;
    return r;
}
static inline int kTilesOf(int K) { return (K + 31) / 32; }
static inline int nTilesOf(int N) { return (N + 15) / 16; }

static void launch_gemm(hipStream_t s, const _Float16* A, const _Float16* Bfrag,
                        const float* bias, _Float16* oh, float* of,
                        int M, int N, int K, int Lout, int rowStride,
                        int batchStride, int act) {
    dim3 grid(nTilesOf(N), (M + 15) / 16);
    dti_wmma_gemm<<<grid, dim3(32), 0, s>>>(A, Bfrag, bias, oh, of, M, N,
                                            kTilesOf(K), Lout, rowStride,
                                            batchStride, act);
}
static void pack_conv(hipStream_t s, const float* w, _Float16* out,
                      int Cout, int Cin, int Kk) {
    int kt = kTilesOf(Kk * Cin), nt = nTilesOf(Cout);
    int tot = kt * nt * 512;
    dti_pack_frag_conv<<<(tot + 255) / 256, 256, 0, s>>>(w, out, Cout, Cin, Kk, kt, nt);
}
static void pack_mat(hipStream_t s, const float* W, _Float16* out, int K, int N) {
    int kt = kTilesOf(K), nt = nTilesOf(N);
    int tot = kt * nt * 512;
    dti_pack_frag_mat<<<(tot + 255) / 256, 256, 0, s>>>(W, out, K, N, kt, nt);
}
static inline size_t fragElems(int K, int N) {
    return (size_t)kTilesOf(K) * nTilesOf(N) * 512;
}

extern "C" void kernel_launch(void* const* d_in, const int* in_sizes, int n_in,
                              void* d_out, int out_size, void* d_ws, size_t ws_size,
                              hipStream_t stream) {
    (void)in_sizes; (void)n_in; (void)out_size; (void)ws_size;
    const int*   peptide = (const int*)d_in[0];
    const int*   MHC     = (const int*)d_in[1];
    const float* pep_emb = (const float*)d_in[2];
    const float* mhc_emb = (const float*)d_in[3];
    const float* pw1 = (const float*)d_in[4];  const float* pb1 = (const float*)d_in[5];
    const float* pw2 = (const float*)d_in[6];  const float* pb2 = (const float*)d_in[7];
    const float* pw3 = (const float*)d_in[8];  const float* pb3 = (const float*)d_in[9];
    const float* mw1 = (const float*)d_in[10]; const float* mb1 = (const float*)d_in[11];
    const float* mw2 = (const float*)d_in[12]; const float* mb2 = (const float*)d_in[13];
    const float* mw3 = (const float*)d_in[14]; const float* mb3 = (const float*)d_in[15];
    const float* Wpa = (const float*)d_in[16]; const float* bpa = (const float*)d_in[17];
    const float* Wma = (const float*)d_in[18]; const float* bma = (const float*)d_in[19];
    const float* Wa  = (const float*)d_in[20]; const float* ba  = (const float*)d_in[21];
    const float* W1  = (const float*)d_in[22]; const float* b1  = (const float*)d_in[23];
    const float* W2  = (const float*)d_in[24]; const float* b2  = (const float*)d_in[25];
    const float* W3  = (const float*)d_in[26]; const float* b3  = (const float*)d_in[27];
    const float* Wo  = (const float*)d_in[28]; const float* bo  = (const float*)d_in[29];
    float* out = (float*)d_out;

    char* wp = (char*)d_ws;
    _Float16* pe   = (_Float16*)wsTake(wp, (size_t)NB * LP0 * DIM, 2);
    _Float16* me   = (_Float16*)wsTake(wp, (size_t)NB * LM0 * DIM, 2);
    _Float16* pw1f = (_Float16*)wsTake(wp, fragElems(4 * DIM, CONV), 2);
    _Float16* pw2f = (_Float16*)wsTake(wp, fragElems(6 * CONV, C2), 2);
    _Float16* pw3f = (_Float16*)wsTake(wp, fragElems(8 * C2, C4), 2);
    _Float16* mw1f = (_Float16*)wsTake(wp, fragElems(4 * DIM, CONV), 2);
    _Float16* mw2f = (_Float16*)wsTake(wp, fragElems(6 * CONV, C2), 2);
    _Float16* mw3f = (_Float16*)wsTake(wp, fragElems(8 * C2, C4), 2);
    _Float16* Wpaf = (_Float16*)wsTake(wp, fragElems(C4, C4), 2);
    _Float16* Wmaf = (_Float16*)wsTake(wp, fragElems(C4, C4), 2);
    _Float16* Waf  = (_Float16*)wsTake(wp, fragElems(C4, C4), 2);
    _Float16* W1f  = (_Float16*)wsTake(wp, fragElems(320, 1024), 2);
    _Float16* W2f  = (_Float16*)wsTake(wp, fragElems(1024, 1024), 2);
    _Float16* W3f  = (_Float16*)wsTake(wp, fragElems(1024, 512), 2);
    _Float16* Wof  = (_Float16*)wsTake(wp, fragElems(512, 2), 2);
    _Float16* p1   = (_Float16*)wsTake(wp, (size_t)NB * LP1 * CONV, 2);
    _Float16* p2   = (_Float16*)wsTake(wp, (size_t)NB * LP2 * C2, 2);
    _Float16* pc   = (_Float16*)wsTake(wp, (size_t)NB * LP3 * C4, 2);
    _Float16* m1   = (_Float16*)wsTake(wp, (size_t)NB * LM1 * CONV, 2);
    _Float16* m2   = (_Float16*)wsTake(wp, (size_t)NB * LM2 * C2, 2);
    _Float16* mc   = (_Float16*)wsTake(wp, (size_t)NB * LM3 * C4, 2);
    float*    pAtt = (float*)wsTake(wp, (size_t)NB * LP3 * C4, 4);
    float*    mAtt = (float*)wsTake(wp, (size_t)NB * LM3 * C4, 4);
    float*    Sp   = (float*)wsTake(wp, (size_t)NB * LP3 * C4, 4);
    float*    Sm   = (float*)wsTake(wp, (size_t)NB * LM3 * C4, 4);
    _Float16* Sph  = (_Float16*)wsTake(wp, (size_t)NB * LP3 * C4, 2);
    _Float16* Smh  = (_Float16*)wsTake(wp, (size_t)NB * LM3 * C4, 2);
    float*    Gp   = (float*)wsTake(wp, (size_t)NB * LP3 * C4, 4);
    float*    Gm   = (float*)wsTake(wp, (size_t)NB * LM3 * C4, 4);
    _Float16* pair = (_Float16*)wsTake(wp, (size_t)NB * 320, 2);
    _Float16* f1   = (_Float16*)wsTake(wp, (size_t)NB * 1024, 2);
    _Float16* f2   = (_Float16*)wsTake(wp, (size_t)NB * 1024, 2);
    _Float16* f3   = (_Float16*)wsTake(wp, (size_t)NB * 512, 2);

    // -- embeddings --------------------------------------------------------
    dti_embed<<<(NB * LP0 * DIM + 255) / 256, 256, 0, stream>>>(peptide, pep_emb, pe, NB * LP0);
    dti_embed<<<(NB * LM0 * DIM + 255) / 256, 256, 0, stream>>>(MHC, mhc_emb, me, NB * LM0);

    // -- weight repack to WMMA fragment order (zero-padded) ---------------
    pack_conv(stream, pw1, pw1f, CONV, DIM, 4);
    pack_conv(stream, pw2, pw2f, C2, CONV, 6);
    pack_conv(stream, pw3, pw3f, C4, C2, 8);
    pack_conv(stream, mw1, mw1f, CONV, DIM, 4);
    pack_conv(stream, mw2, mw2f, C2, CONV, 6);
    pack_conv(stream, mw3, mw3f, C4, C2, 8);
    pack_mat(stream, Wpa, Wpaf, C4, C4);
    pack_mat(stream, Wma, Wmaf, C4, C4);
    pack_mat(stream, Wa, Waf, C4, C4);
    pack_mat(stream, W1, W1f, 320, 1024);
    pack_mat(stream, W2, W2f, 1024, 1024);
    pack_mat(stream, W3, W3f, 1024, 512);
    pack_mat(stream, Wo, Wof, 512, 2);

    // -- conv stacks as implicit GEMM (relu, f16 out) ---------------------
    launch_gemm(stream, pe, pw1f, pb1, p1, nullptr, NB * LP1, CONV, 4 * DIM, LP1, DIM, LP0 * DIM, 1);
    launch_gemm(stream, p1, pw2f, pb2, p2, nullptr, NB * LP2, C2, 6 * CONV, LP2, CONV, LP1 * CONV, 1);
    launch_gemm(stream, p2, pw3f, pb3, pc, nullptr, NB * LP3, C4, 8 * C2, LP3, C2, LP2 * C2, 1);
    launch_gemm(stream, me, mw1f, mb1, m1, nullptr, NB * LM1, CONV, 4 * DIM, LM1, DIM, LM0 * DIM, 1);
    launch_gemm(stream, m1, mw2f, mb2, m2, nullptr, NB * LM2, C2, 6 * CONV, LM2, CONV, LM1 * CONV, 1);
    launch_gemm(stream, m2, mw3f, mb3, mc, nullptr, NB * LM3, C4, 8 * C2, LM3, C2, LM2 * C2, 1);

    // -- attention linears (f32 out, no act) ------------------------------
    launch_gemm(stream, pc, Wpaf, bpa, nullptr, pAtt, NB * LP3, C4, C4, NB * LP3, C4, 0, 0);
    launch_gemm(stream, mc, Wmaf, bma, nullptr, mAtt, NB * LM3, C4, C4, NB * LM3, C4, 0, 0);

    // -- cross relu-means (mean commuted through Wa) ----------------------
    dti_cross_relu_mean<<<dim3(LP3, NB), 160, 0, stream>>>(pAtt, mAtt, Sp, LP3, LM3, C4);
    dti_cross_relu_mean<<<dim3(LM3, NB), 160, 0, stream>>>(mAtt, pAtt, Sm, LM3, LP3, C4);
    dti_cvt_h<<<(NB * LP3 * C4 + 255) / 256, 256, 0, stream>>>(Sp, Sph, (long)NB * LP3 * C4);
    dti_cvt_h<<<(NB * LM3 * C4 + 255) / 256, 256, 0, stream>>>(Sm, Smh, (long)NB * LM3 * C4);

    // -- gate pre-activations: G = S @ Wa + ba ----------------------------
    launch_gemm(stream, Sph, Waf, ba, nullptr, Gp, NB * LP3, C4, C4, NB * LP3, C4, 0, 0);
    launch_gemm(stream, Smh, Waf, ba, nullptr, Gm, NB * LM3, C4, C4, NB * LM3, C4, 0, 0);

    // -- sigmoid gate + global max pool -> pair [8,320] -------------------
    dti_gate_max<<<NB, 160, 0, stream>>>(pc, Gp, pair, LP3, C4, 320, 0);
    dti_gate_max<<<NB, 160, 0, stream>>>(mc, Gm, pair, LM3, C4, 320, C4);

    // -- FC head ----------------------------------------------------------
    launch_gemm(stream, pair, W1f, b1, f1, nullptr, NB, 1024, 320, NB, 320, 0, 2);
    launch_gemm(stream, f1, W2f, b2, f2, nullptr, NB, 1024, 1024, NB, 1024, 0, 2);
    launch_gemm(stream, f2, W3f, b3, f3, nullptr, NB, 512, 1024, NB, 1024, 0, 2);
    launch_gemm(stream, f3, Wof, bo, nullptr, out, NB, 2, 512, NB, 512, 0, 0);
}
